// TransformerBlock_55241869361251
// MI455X (gfx1250) — matrix operand order repaired and checked by reading, not compiled, not run on hardware
//
#include <hip/hip_runtime.h>

// ---------------------------------------------------------------------------
// MI455X (gfx1250) transformer block.
//   * fp32 inputs -> bf16 operands, fp32 accumulate via v_wmma_f32_16x16x32_bf16
//   * all five GEMMs + both attention matmuls on the matrix cores
//   * weights pre-transposed once so GEMM tiles are pure 2-D copies, staged
//     into LDS by the Tensor Data Mover (tensor_load_to_lds, TENSORcnt),
//     double-buffered: TDM for tile i+1 overlaps WMMA on tile i
//   * flash-style streaming softmax (scores never hit HBM)
// ---------------------------------------------------------------------------

typedef __attribute__((ext_vector_type(16))) __bf16       v16bf;
typedef __attribute__((ext_vector_type(8)))  float        v8f;
typedef __attribute__((ext_vector_type(4)))  unsigned int u32x4;
typedef __attribute__((ext_vector_type(4)))  int          i32x4;
typedef __attribute__((ext_vector_type(8)))  int          i32x8;

#define ZERO8 {0.f,0.f,0.f,0.f,0.f,0.f,0.f,0.f}

static __device__ __forceinline__ unsigned short f2bf(float f) {
  unsigned int u = __float_as_uint(f);
  u += 0x7fffu + ((u >> 16) & 1u);   // round-to-nearest-even
  return (unsigned short)(u >> 16);
}

static __device__ __forceinline__ v8f wmma_bf16(v16bf a, v16bf b, v8f c) {
  return __builtin_amdgcn_wmma_f32_16x16x32_bf16(
      false, a, false, b, (short)0, c, false, false);
}

// Assemble a v16bf fragment from two 16-byte runs (slots 0..7 and 8..15).
static __device__ __forceinline__ v16bf frag16(const unsigned short* p0,
                                               const unsigned short* p1) {
  union { u32x4 u[2]; v16bf v; } t;
  t.u[0] = *(const u32x4*)p0;
  t.u[1] = *(const u32x4*)p1;
  return t.v;
}

static __device__ __forceinline__ float halfmax(float v) {
  v = fmaxf(v, __shfl_xor(v, 1, 32));
  v = fmaxf(v, __shfl_xor(v, 2, 32));
  v = fmaxf(v, __shfl_xor(v, 4, 32));
  v = fmaxf(v, __shfl_xor(v, 8, 32));
  return v;
}
static __device__ __forceinline__ float halfsum(float v) {
  v += __shfl_xor(v, 1, 32);
  v += __shfl_xor(v, 2, 32);
  v += __shfl_xor(v, 4, 32);
  v += __shfl_xor(v, 8, 32);
  return v;
}

static __device__ __forceinline__ float gelu_exact(float x) {
  return 0.5f * x * (1.0f + erff(x * 0.70710678118654752f));
}

// ---------------------------------------------------------------------------
// Tensor Data Mover: async copy of a 2-D tile (64 rows x 64 bytes) from
// global into LDS.  D# packed per CDNA5 ISA ch.8: data_size=8B, tile_dim0=8
// units (64B), tile_dim1=64 rows, tensor_dim0_stride = row pitch.  Issued by
// one wave; completion via s_wait_tensorcnt + workgroup barrier.
// ---------------------------------------------------------------------------
static __device__ __forceinline__ void tdm_load_64x64B(
    unsigned lds_addr, const unsigned short* gptr, unsigned row_pitch_bytes) {
  const unsigned long long ga = (unsigned long long)(size_t)gptr;
  const unsigned stride_u = row_pitch_bytes >> 3;          // 8-byte units
  i32x4 g0;
  g0[0] = 1;                                               // count=1 (valid)
  g0[1] = (int)lds_addr;                                   // LDS byte address
  g0[2] = (int)(unsigned)(ga & 0xFFFFFFFFull);             // global_addr[31:0]
  g0[3] = (int)(((unsigned)(ga >> 32) & 0x01FFFFFFu) | (2u << 30)); // [56:32]|type=2
  i32x8 g1;
  const unsigned td0 = stride_u;                           // tensor_dim0 (OOB)
  const unsigned td1 = 1u << 20;                           // tensor_dim1 (OOB)
  g1[0] = (int)(3u << 16);                                 // data_size = 8B
  g1[1] = (int)((td0 & 0xFFFFu) << 16);                    // tensor_dim0[15:0]
  g1[2] = (int)((td0 >> 16) | ((td1 & 0xFFFFu) << 16));    // td0[31:16], td1[15:0]
  g1[3] = (int)(((td1 >> 16) & 0xFFFFu) | (8u << 16));     // td1[31:16], tile_dim0=8
  g1[4] = (int)64u;                                        // tile_dim1=64, tile_dim2=0
  g1[5] = (int)stride_u;                                   // tensor_dim0_stride[31:0]
  g1[6] = 0;                                               // stride0[47:32], stride1 lo
  g1[7] = 0;
  i32x4 z = {0, 0, 0, 0};
  asm volatile("tensor_load_to_lds %0, %1, %2, %3"
               :
               : "s"(g0), "s"(g1), "s"(z), "s"(z)
               : "memory");
}

// ---------------------------------------------------------------------------
// fp32 (K x N) -> bf16 transposed (N x K).  One-time weight prep so GEMM
// B-tiles are row-major 2-D copies (TDM-friendly, fast b128 fragment reads).
// ---------------------------------------------------------------------------
__global__ __launch_bounds__(256)
void cvt_t_bf16_kernel(const float* __restrict__ in,
                       unsigned short* __restrict__ out, int K, int N) {
  __shared__ unsigned short tile[32][33];
  const int kb = blockIdx.x * 32, nb = blockIdx.y * 32;
  const int tx = threadIdx.x & 31, ty = threadIdx.x >> 5;
  #pragma unroll
  for (int r = ty; r < 32; r += 8)
    tile[r][tx] = f2bf(in[(size_t)(kb + r) * N + nb + tx]);
  __syncthreads();
  #pragma unroll
  for (int r = ty; r < 32; r += 8)
    out[(size_t)(nb + r) * K + kb + tx] = tile[tx][r];
}

// ---------------------------------------------------------------------------
// RMSNorm: one row (1024 elems) per block.  out_f32 optional (nullable).
// ---------------------------------------------------------------------------
__global__ __launch_bounds__(256)
void rmsnorm_kernel(const float* __restrict__ x, const float* __restrict__ gain,
                    unsigned short* __restrict__ out_bf,
                    float* __restrict__ out_f32) {
  __shared__ float red[8];
  const int row = blockIdx.x;
  const float* xr = x + (size_t)row * 1024;
  float s = 0.f;
  for (int c = threadIdx.x; c < 1024; c += 256) { float v = xr[c]; s += v * v; }
  #pragma unroll
  for (int m = 1; m < 32; m <<= 1) s += __shfl_xor(s, m, 32);
  if ((threadIdx.x & 31) == 0) red[threadIdx.x >> 5] = s;
  __syncthreads();
  float ms = 0.f;
  #pragma unroll
  for (int i = 0; i < 8; i++) ms += red[i];
  ms *= (1.0f / 1024.0f);
  const float sc = rsqrtf(ms + 1e-8f);
  unsigned short* ob = out_bf + (size_t)row * 1024;
  for (int c = threadIdx.x; c < 1024; c += 256) {
    float v = xr[c] * sc * gain[c];
    ob[c] = f2bf(v);
    if (out_f32) out_f32[(size_t)row * 1024 + c] = v;
  }
}

// ---------------------------------------------------------------------------
// GEMM: C(MxN) = A(MxK,bf16) * Bt(NxK,bf16)^T.  Block tile 64x64, 4 waves,
// each wave -> 16(M) x 64(N) = 4 WMMA accumulators.  K step 32.
// Tiles staged by TDM, double-buffered (one tile-set in flight).
// EP==0: store bf16.  EP==1: store fp32 acc + resid.
// ---------------------------------------------------------------------------
template <int EP>
__global__ __launch_bounds__(128)
void gemm_bf16(const unsigned short* __restrict__ A,
               const unsigned short* __restrict__ Bt,
               const float* __restrict__ resid,
               void* __restrict__ out, int M, int N, int K) {
  __shared__ unsigned short Alds[2][64 * 32];
  __shared__ unsigned short Blds[2][64 * 32];
  const int t    = threadIdx.x;
  const int lane = t & 31, wave = t >> 5;
  const int hi   = lane >> 4, l16 = lane & 15;
  const int kb8  = hi * 8,  kb16 = hi * 16;
  const int mTile = blockIdx.y * 64;
  const int nTile = blockIdx.x * 64;
  const bool issuer = (wave == 0);

  const unsigned aoff = (unsigned)(size_t)&Alds[0][0];
  const unsigned boff = (unsigned)(size_t)&Blds[0][0];
  const unsigned short* Ag = A  + (size_t)mTile * K;
  const unsigned short* Bg = Bt + (size_t)nTile * K;

  v8f acc[4];
  #pragma unroll
  for (int i = 0; i < 4; i++) { v8f z = ZERO8; acc[i] = z; }

  const int nk = K >> 5;
  if (issuer) {
    tdm_load_64x64B(aoff, Ag, (unsigned)K * 2);
    tdm_load_64x64B(boff, Bg, (unsigned)K * 2);
  }
  for (int i = 0; i < nk; i++) {
    const int cur = i & 1;
    if (issuer) {
      if (i + 1 < nk) {
        tdm_load_64x64B(aoff + (1 - cur) * 4096, Ag + (i + 1) * 32, (unsigned)K * 2);
        tdm_load_64x64B(boff + (1 - cur) * 4096, Bg + (i + 1) * 32, (unsigned)K * 2);
        __builtin_amdgcn_s_wait_tensorcnt(2);   // next tile-set may stay in flight
      } else {
        __builtin_amdgcn_s_wait_tensorcnt(0);   // drain before last compute
      }
    }
    __syncthreads();

    const unsigned short* al = &Alds[cur][0];
    const unsigned short* bl = &Blds[cur][0];
    const int mw = wave * 16 + l16;
    v16bf af = frag16(al + mw * 32 + kb16, al + mw * 32 + kb16 + 8);
    v16bf bf0 = frag16(bl + (0 * 16 + l16) * 32 + kb16, bl + (0 * 16 + l16) * 32 + kb16 + 8);
    v16bf bf1 = frag16(bl + (1 * 16 + l16) * 32 + kb16, bl + (1 * 16 + l16) * 32 + kb16 + 8);
    v16bf bf2 = frag16(bl + (2 * 16 + l16) * 32 + kb16, bl + (2 * 16 + l16) * 32 + kb16 + 8);
    v16bf bf3 = frag16(bl + (3 * 16 + l16) * 32 + kb16, bl + (3 * 16 + l16) * 32 + kb16 + 8);
    acc[0] = wmma_bf16(af, bf0, acc[0]);
    acc[1] = wmma_bf16(af, bf1, acc[1]);
    acc[2] = wmma_bf16(af, bf2, acc[2]);
    acc[3] = wmma_bf16(af, bf3, acc[3]);
    __syncthreads();
  }

  #pragma unroll
  for (int i = 0; i < 4; i++) {
    #pragma unroll
    for (int r = 0; r < 8; r++) {
      int m = mTile + wave * 16 + hi * 8 + r;
      int n = nTile + i * 16 + l16;
      size_t idx = (size_t)m * N + n;
      float v = acc[i][r];
      if (EP == 0)      ((unsigned short*)out)[idx] = f2bf(v);
      else              ((float*)out)[idx] = v + resid[idx];
    }
  }
}

// ---------------------------------------------------------------------------
// Dual GEMM (MLP front): out = gelu(A*B1t^T) + A*B2t^T, stored bf16.
// Shared A fragment; three TDM streams per k-step.
// ---------------------------------------------------------------------------
__global__ __launch_bounds__(128)
void gemm_dual_bf16(const unsigned short* __restrict__ A,
                    const unsigned short* __restrict__ B1t,
                    const unsigned short* __restrict__ B2t,
                    unsigned short* __restrict__ out, int M, int N, int K) {
  __shared__ unsigned short Alds[2][64 * 32];
  __shared__ unsigned short B1lds[2][64 * 32];
  __shared__ unsigned short B2lds[2][64 * 32];
  const int t    = threadIdx.x;
  const int lane = t & 31, wave = t >> 5;
  const int hi   = lane >> 4, l16 = lane & 15;
  const int kb8  = hi * 8,  kb16 = hi * 16;
  const int mTile = blockIdx.y * 64;
  const int nTile = blockIdx.x * 64;
  const bool issuer = (wave == 0);

  const unsigned aoff  = (unsigned)(size_t)&Alds[0][0];
  const unsigned b1off = (unsigned)(size_t)&B1lds[0][0];
  const unsigned b2off = (unsigned)(size_t)&B2lds[0][0];
  const unsigned short* Ag  = A   + (size_t)mTile * K;
  const unsigned short* B1g = B1t + (size_t)nTile * K;
  const unsigned short* B2g = B2t + (size_t)nTile * K;

  v8f acc1[4], acc2[4];
  #pragma unroll
  for (int i = 0; i < 4; i++) { v8f z = ZERO8; acc1[i] = z; acc2[i] = z; }

  const int nk = K >> 5;
  if (issuer) {
    tdm_load_64x64B(aoff,  Ag,  (unsigned)K * 2);
    tdm_load_64x64B(b1off, B1g, (unsigned)K * 2);
    tdm_load_64x64B(b2off, B2g, (unsigned)K * 2);
  }
  for (int i = 0; i < nk; i++) {
    const int cur = i & 1;
    if (issuer) {
      if (i + 1 < nk) {
        tdm_load_64x64B(aoff  + (1 - cur) * 4096, Ag  + (i + 1) * 32, (unsigned)K * 2);
        tdm_load_64x64B(b1off + (1 - cur) * 4096, B1g + (i + 1) * 32, (unsigned)K * 2);
        tdm_load_64x64B(b2off + (1 - cur) * 4096, B2g + (i + 1) * 32, (unsigned)K * 2);
        __builtin_amdgcn_s_wait_tensorcnt(3);   // next triple may stay in flight
      } else {
        __builtin_amdgcn_s_wait_tensorcnt(0);
      }
    }
    __syncthreads();

    const unsigned short* al  = &Alds[cur][0];
    const unsigned short* b1l = &B1lds[cur][0];
    const unsigned short* b2l = &B2lds[cur][0];
    const int mw = wave * 16 + l16;
    v16bf af = frag16(al + mw * 32 + kb16, al + mw * 32 + kb16 + 8);
    #pragma unroll
    for (int i2 = 0; i2 < 4; i2++) {
      const unsigned short* p1 = b1l + (i2 * 16 + l16) * 32 + kb16;
      const unsigned short* p2 = b2l + (i2 * 16 + l16) * 32 + kb16;
      acc1[i2] = wmma_bf16(af, frag16(p1, p1 + 8), acc1[i2]);
      acc2[i2] = wmma_bf16(af, frag16(p2, p2 + 8), acc2[i2]);
    }
    __syncthreads();
  }

  #pragma unroll
  for (int i = 0; i < 4; i++) {
    #pragma unroll
    for (int r = 0; r < 8; r++) {
      int m = mTile + wave * 16 + hi * 8 + r;
      int n = nTile + i * 16 + l16;
      out[(size_t)m * N + n] = f2bf(gelu_exact(acc1[i][r]) + acc2[i][r]);
    }
  }
}

// ---------------------------------------------------------------------------
// Flash attention (causal, GQA).  One wave = 16 query rows; S streamed in
// steps of 32.  Per step: 4 WMMAs Q*K^T + 4 WMMAs P*V, online softmax with
// intra-half shuffles; next strip prefetched (global_prefetch_b8).
// q layout: (B*T, 1024) bf16; k,v: (B*T, 256) bf16; o: (B*T, 1024) bf16.
// ---------------------------------------------------------------------------
__global__ __launch_bounds__(128)
void attn_kernel(const unsigned short* __restrict__ q,
                 const unsigned short* __restrict__ k,
                 const unsigned short* __restrict__ v,
                 unsigned short* __restrict__ o) {
  __shared__ unsigned short Plds[4][16 * 32];   // per-wave P (C->A relayout)
  __shared__ unsigned short Vt[4][64 * 32];     // per-wave V transposed [h][s]
  const int t    = threadIdx.x;
  const int lane = t & 31, wave = t >> 5;
  const int hi   = lane >> 4, l16 = lane & 15;
  const int kb8  = hi * 8,  kb16 = hi * 16;
  const int b    = blockIdx.x >> 4;       // 16 q heads
  const int qh   = blockIdx.x & 15;
  const int kvh  = qh >> 2;               // G = 4
  const int q0   = blockIdx.y * 64 + wave * 16;

  const size_t qrow = ((size_t)(b * 2048 + q0 + l16)) * 1024 + qh * 64;
  v16bf qf0 = frag16(q + qrow + kb16,      q + qrow + kb16 + 8);
  v16bf qf1 = frag16(q + qrow + 32 + kb16, q + qrow + 32 + kb16 + 8);

  float mrow[8], lrow[8];
  v8f oacc[4];
  #pragma unroll
  for (int i = 0; i < 4; i++) { v8f z = ZERO8; oacc[i] = z; }
  #pragma unroll
  for (int r = 0; r < 8; r++) { mrow[r] = -__builtin_inff(); lrow[r] = 0.f; }

  const int nSteps = (q0 + 47) >> 5;   // cover keys 0 .. q0+15
  for (int st = 0; st < nSteps; ++st) {
    const int s0 = st * 32;

    if (st + 1 < nSteps) {             // prefetch next K/V strip
      __builtin_prefetch(k + ((size_t)(b * 2048 + s0 + 32 + l16)) * 256 + kvh * 64, 0, 3);
      __builtin_prefetch(v + ((size_t)(b * 2048 + s0 + 32 + lane)) * 256 + kvh * 64, 0, 3);
    }

    // ---- scores: two 16x16 tiles over this 32-key strip ----
    v8f sc[2];
    #pragma unroll
    for (int tt = 0; tt < 2; ++tt) {
      size_t krow = ((size_t)(b * 2048 + s0 + tt * 16 + l16)) * 256 + kvh * 64;
      v16bf kf0 = frag16(k + krow + kb16,      k + krow + kb16 + 8);
      v16bf kf1 = frag16(k + krow + 32 + kb16, k + krow + 32 + kb16 + 8);
      v8f z = ZERO8;
      z = wmma_bf16(qf0, kf0, z);
      z = wmma_bf16(qf1, kf1, z);
      sc[tt] = z;
    }

    // ---- stage V strip transposed: lane loads one key row (64 bf16) ----
    {
      size_t vrow = ((size_t)(b * 2048 + s0 + lane)) * 256 + kvh * 64;
      union { u32x4 u[8]; unsigned short s[64]; } vv;
      #pragma unroll
      for (int c = 0; c < 8; c++) vv.u[c] = *(const u32x4*)(v + vrow + c * 8);
      #pragma unroll
      for (int h = 0; h < 64; h++) Vt[wave][h * 32 + lane] = vv.s[h];
    }

    // ---- causal mask + scale + online softmax ----
    float p0[8], p1[8], alpha[8];
    #pragma unroll
    for (int r = 0; r < 8; r++) {
      int qi  = q0 + hi * 8 + r;
      float x0 = (s0 + l16      <= qi) ? sc[0][r] * 0.125f : -__builtin_inff();
      float x1 = (s0 + 16 + l16 <= qi) ? sc[1][r] * 0.125f : -__builtin_inff();
      float mx   = halfmax(fmaxf(x0, x1));
      float mnew = fmaxf(mrow[r], mx);
      p0[r] = __expf(x0 - mnew);
      p1[r] = __expf(x1 - mnew);
      float rs = halfsum(p0[r] + p1[r]);
      alpha[r] = __expf(mrow[r] - mnew);
      lrow[r]  = lrow[r] * alpha[r] + rs;
      mrow[r]  = mnew;
    }

    // ---- relayout P (C layout -> A operand layout) via per-wave LDS ----
    #pragma unroll
    for (int r = 0; r < 8; r++) {
      int Mr = hi * 8 + r;
      Plds[wave][Mr * 32 + l16]      = f2bf(p0[r]);
      Plds[wave][Mr * 32 + 16 + l16] = f2bf(p1[r]);
    }
    v16bf pf = frag16(&Plds[wave][l16 * 32 + kb16],
                      &Plds[wave][l16 * 32 + kb16 + 8]);

    // ---- rescale accumulators and P@V (K=32, full) ----
    #pragma unroll
    for (int i = 0; i < 4; i++) {
      #pragma unroll
      for (int r = 0; r < 8; r++) oacc[i][r] *= alpha[r];
      const unsigned short* vp = &Vt[wave][(i * 16 + l16) * 32 + kb16];
      oacc[i] = wmma_bf16(pf, frag16(vp, vp + 8), oacc[i]);
    }
  }

  // ---- normalize and store bf16 ----
  #pragma unroll
  for (int i = 0; i < 4; i++) {
    #pragma unroll
    for (int r = 0; r < 8; r++) {
      size_t idx = ((size_t)(b * 2048 + q0 + hi * 8 + r)) * 1024
                 + qh * 64 + i * 16 + l16;
      o[idx] = f2bf(oacc[i][r] / lrow[r]);
    }
  }
}

// ---------------------------------------------------------------------------
// Host orchestration
// ---------------------------------------------------------------------------
extern "C" void kernel_launch(void* const* d_in, const int* in_sizes, int n_in,
                              void* d_out, int out_size, void* d_ws, size_t ws_size,
                              hipStream_t stream) {
  (void)in_sizes; (void)n_in; (void)out_size; (void)ws_size;
  const int M = 4096, D = 1024, FF = 4096, DKV = 256;

  const float* x     = (const float*)d_in[0];
  const float* Wq    = (const float*)d_in[1];
  const float* Wk    = (const float*)d_in[2];
  const float* Wv    = (const float*)d_in[3];
  const float* Wo    = (const float*)d_in[4];
  const float* g_att = (const float*)d_in[5];
  const float* g_mlp = (const float*)d_in[6];
  const float* Win1  = (const float*)d_in[7];
  const float* Win2  = (const float*)d_in[8];
  const float* Wout  = (const float*)d_in[9];
  float* out = (float*)d_out;

  size_t off = 0;
  char* base = (char*)d_ws;
  auto alloc = [&](size_t bytes) -> void* {
    void* p = base + off;
    off += (bytes + 255) & ~(size_t)255;
    return p;
  };
  unsigned short* xn   = (unsigned short*)alloc((size_t)M * D * 2);
  unsigned short* wqT  = (unsigned short*)alloc((size_t)D * D * 2);     // (N,K)
  unsigned short* wkT  = (unsigned short*)alloc((size_t)DKV * D * 2);
  unsigned short* wvT  = (unsigned short*)alloc((size_t)DKV * D * 2);
  unsigned short* woT  = (unsigned short*)alloc((size_t)D * D * 2);
  unsigned short* w1T  = (unsigned short*)alloc((size_t)FF * D * 2);
  unsigned short* w2T  = (unsigned short*)alloc((size_t)FF * D * 2);
  unsigned short* woutT= (unsigned short*)alloc((size_t)D * FF * 2);
  unsigned short* qb   = (unsigned short*)alloc((size_t)M * D * 2);
  unsigned short* kbuf = (unsigned short*)alloc((size_t)M * DKV * 2);
  unsigned short* vbuf = (unsigned short*)alloc((size_t)M * DKV * 2);
  unsigned short* ob   = (unsigned short*)alloc((size_t)M * D * 2);
  float*          hf   = (float*)alloc((size_t)M * D * 4);
  unsigned short* hnb  = (unsigned short*)alloc((size_t)M * D * 2);
  float*          hnf  = (float*)alloc((size_t)M * D * 4);
  unsigned short* hmid = (unsigned short*)alloc((size_t)M * FF * 2);

  // One-time: fp32 (KxN) weights -> bf16 transposed (NxK)
  auto cvtT = [&](const float* src, unsigned short* dst, int K, int N) {
    cvt_t_bf16_kernel<<<dim3(K / 32, N / 32), 256, 0, stream>>>(src, dst, K, N);
  };
  cvtT(Wq,   wqT,   D, D);
  cvtT(Wk,   wkT,   D, DKV);
  cvtT(Wv,   wvT,   D, DKV);
  cvtT(Wo,   woT,   D, D);
  cvtT(Win1, w1T,   D, FF);
  cvtT(Win2, w2T,   D, FF);
  cvtT(Wout, woutT, FF, D);

  // 1. xn = rmsnorm(x) * gain_attn  (bf16)
  rmsnorm_kernel<<<M, 256, 0, stream>>>(x, g_att, xn, nullptr);

  // 2. Q/K/V projections (WMMA GEMM, TDM-staged, bf16 out)
  gemm_bf16<0><<<dim3(D / 64, M / 64), 128, 0, stream>>>(xn, wqT, nullptr, qb, M, D, D);
  gemm_bf16<0><<<dim3(DKV / 64, M / 64), 128, 0, stream>>>(xn, wkT, nullptr, kbuf, M, DKV, D);
  gemm_bf16<0><<<dim3(DKV / 64, M / 64), 128, 0, stream>>>(xn, wvT, nullptr, vbuf, M, DKV, D);

  // 3. causal GQA flash attention
  attn_kernel<<<dim3(32, 32), 128, 0, stream>>>(qb, kbuf, vbuf, ob);

  // 4. output projection + residual -> h (fp32)
  gemm_bf16<1><<<dim3(D / 64, M / 64), 128, 0, stream>>>(ob, woT, x, hf, M, D, D);

  // 5. hn = rmsnorm(h) * gain_mlp  (bf16 + fp32: also the final residual)
  rmsnorm_kernel<<<M, 256, 0, stream>>>(hf, g_mlp, hnb, hnf);

  // 6. hmid = gelu(hn @ W_in1) + hn @ W_in2  (dual GEMM, bf16)
  gemm_dual_bf16<<<dim3(FF / 64, M / 64), 128, 0, stream>>>(hnb, w1T, w2T, hmid, M, FF, D);

  // 7. out = hn + hmid @ W_out  (fp32)
  gemm_bf16<1><<<dim3(D / 64, M / 64), 128, 0, stream>>>(hmid, woutT, hnf, out, M, D, FF);
}
